// Note_VQVAE_43387759624353
// MI455X (gfx1250) — matrix operand order repaired and checked by reading, not compiled, not run on hardware
//
#include <hip/hip_runtime.h>
#include <math.h>

// ---------------------------------------------------------------------------
// CDNA5 (gfx1250) implementation of the Note-VQVAE forward pass.
// All dense math runs through v_wmma_f32_16x16x32_f16 (f32 accumulate).
// GEMM: 64x64 block tile (4 waves, 2x2 WMMA micro-tiles per wave), weight
// tile double-buffered in LDS via global_load_async_to_lds_b128 + 
// s_wait_asynccnt, BN+leaky fused into conv epilogues, global_prefetch on A.
// ---------------------------------------------------------------------------

typedef _Float16 v16h __attribute__((ext_vector_type(16)));
typedef float    v8f  __attribute__((ext_vector_type(8)));

// ---------------------------------------------------------------------------
// Generic WMMA GEMM:  C[M,N] = A(M,K) @ W(N,K)^T + bias  [+ BN + leaky 0.2]
//   AMODE 0: A row-major with lda
//   AMODE 1: A = im2col of conv1d input (B,Cin,Tin), stride/pad, row=(b,tout)
//   AMODE 2: A = im2col of convT input (lhs-dilated by `stride`, left pad `pad`)
//   SMODE 0: C[row*N + n]
//   SMODE 1: C[((b*N)+n)*Tout + t]   (channel-major conv output)
//   EPI   1: y = (acc+bias)*scale[n] + shift[n]; leaky(0.2)   (inference BN)
//   ASYNCB1: stage W tile (64x32 f32) in LDS via async loads (needs K%32==0)
// Requires M%64==0 and N%64==0 (true for every call site in this model), so
// every block/wave is fully active and EXEC is all-ones for WMMA.
// A/B k order is contiguous (k = 16*(lane/16) + e) in both, not the ISA table order; C/D map is the ISA's.
// ---------------------------------------------------------------------------
template <int AMODE, int SMODE, int EPI, int ASYNCB>
__global__ void __launch_bounds__(128)
wmma_gemm_kernel(const float* __restrict__ A, const float* __restrict__ W,
                 const float* __restrict__ bias, float* __restrict__ C,
                 const float* __restrict__ bng, const float* __restrict__ bnb,
                 const float* __restrict__ bnm, const float* __restrict__ bnv,
                 int M, int N, int K, int lda,
                 int Cin, int KW, int stride, int pad, int Tin, int Tout)
{
  __shared__ float bsh[ASYNCB ? 2 * 64 * 32 : 1];  // double-buffered W tile

  const int tid  = threadIdx.x;
  const int lane = tid & 31;
  const int wv   = tid >> 5;
  const int nbn  = N >> 6;
  const int bm   = blockIdx.x / nbn;
  const int bn   = blockIdx.x - bm * nbn;
  const int wm   = wv >> 1;        // 2x2 wave grid over the 64x64 block tile
  const int wn   = wv & 1;

  const int l15    = lane & 15;
  const int arow0  = bm * 64 + wm * 32 + l15;
  const int arow1  = arow0 + 16;
  const int bcol0g = bn * 64 + wn * 32 + l15;   // global B col (direct/bias)
  const int bcol1g = bcol0g + 16;
  const int bloc0  = wn * 32 + l15;             // local B col inside LDS tile
  const int bloc1  = bloc0 + 16;
  const int kofsA  = (lane < 16) ? 0 : 16;
  const int kofsB  = (lane < 16) ? 0 : 16;

  int bb0 = 0, tt0 = 0, bb1 = 0, tt1 = 0;
  if constexpr (AMODE != 0) {
    bb0 = arow0 / Tout; tt0 = arow0 - bb0 * Tout;
    bb1 = arow1 / Tout; tt1 = arow1 - bb1 * Tout;
  }

  // A-element gather for one (row, k)
  auto loadA = [&](int bb, int tt, int arow, int kk) -> float {
    float x = 0.f;
    if (kk < K) {
      if constexpr (AMODE == 0) {
        x = A[(size_t)arow * lda + kk];
      } else if constexpr (AMODE == 1) {
        const int ci = kk / KW, kw = kk - ci * KW;
        const int p  = tt * stride - pad + kw;
        if (p >= 0 && p < Tin) x = A[((size_t)bb * Cin + ci) * Tin + p];
      } else {  // transposed conv: gather through lhs dilation
        const int ci = kk / KW, kw = kk - ci * KW;
        const int p  = tt + kw - pad;
        const int dl = (Tin - 1) * stride + 1;
        if (p >= 0 && p < dl && (p % stride) == 0)
          x = A[((size_t)bb * Cin + ci) * Tin + p / stride];
      }
    }
    return x;
  };

  // async stage of W[bn*64 .. +63][k0 .. k0+31] into bsh[buf]
  // thread -> (col = tid>>1, khalf = tid&1): 16 consecutive f32 = 4x b128
  const int      scol  = tid >> 1;
  const int      shalf = tid & 1;
  const uint32_t ldsb  = (uint32_t)(uintptr_t)(void*)&bsh[0];  // LDS byte offset
  auto stage = [&](int buf, int k0) {
    if constexpr (ASYNCB) {
      const float*   g = &W[(size_t)(bn * 64 + scol) * K + k0 + shalf * 16];
      const uint32_t l = ldsb + (uint32_t)(buf * 2048 + scol * 32 + shalf * 16) * 4u;
      asm volatile(
          "global_load_async_to_lds_b128 %0, %1, off\n\t"
          "global_load_async_to_lds_b128 %2, %3, off\n\t"
          "global_load_async_to_lds_b128 %4, %5, off\n\t"
          "global_load_async_to_lds_b128 %6, %7, off"
          :
          : "v"(l), "v"(g), "v"(l + 16u), "v"(g + 4), "v"(l + 32u), "v"(g + 8),
            "v"(l + 48u), "v"(g + 12)
          : "memory");
    }
  };

  if constexpr (ASYNCB) stage(0, 0);

  v8f a00 = {}, a01 = {}, a10 = {}, a11 = {};
  int buf = 0;
  for (int k0 = 0; k0 < K; k0 += 32) {
    if constexpr (ASYNCB) {
      asm volatile("s_wait_asynccnt 0x0" ::: "memory");
      __syncthreads();                    // tile visible to all waves; prior
      if (k0 + 32 < K) stage(buf ^ 1, k0 + 32);  // reads of buf^1 are done
    }

    v16h av0, av1, bv0, bv1;
#pragma unroll
    for (int e = 0; e < 16; ++e) {
      const int kk = k0 + kofsA + e;
      av0[e] = (_Float16)loadA(bb0, tt0, arow0, kk);
      av1[e] = (_Float16)loadA(bb1, tt1, arow1, kk);
    }
    if constexpr (ASYNCB) {
      const float* t0 = &bsh[buf * 2048 + bloc0 * 32 + kofsB];
      const float* t1 = &bsh[buf * 2048 + bloc1 * 32 + kofsB];
#pragma unroll
      for (int e = 0; e < 16; ++e) {      // 16 contiguous f32 -> ds_load_b128s
        bv0[e] = (_Float16)t0[e];
        bv1[e] = (_Float16)t1[e];
      }
    } else {
#pragma unroll
      for (int e = 0; e < 16; ++e) {
        const int  kk = k0 + kofsB + e;
        const bool ok = kk < K;
        bv0[e] = (_Float16)(ok ? W[(size_t)bcol0g * K + kk] : 0.f);
        bv1[e] = (_Float16)(ok ? W[(size_t)bcol1g * K + kk] : 0.f);
      }
    }
    if constexpr (AMODE == 0) {
      if (k0 + 32 < K)  // prefetch next A strip (global_prefetch_b8)
        __builtin_prefetch(&A[(size_t)arow0 * lda + k0 + 32 + kofsA], 0, 1);
    }
    a00 = __builtin_amdgcn_wmma_f32_16x16x32_f16(false, av0, false, bv0, (short)0, a00, false, false);
    a01 = __builtin_amdgcn_wmma_f32_16x16x32_f16(false, av0, false, bv1, (short)0, a01, false, false);
    a10 = __builtin_amdgcn_wmma_f32_16x16x32_f16(false, av1, false, bv0, (short)0, a10, false, false);
    a11 = __builtin_amdgcn_wmma_f32_16x16x32_f16(false, av1, false, bv1, (short)0, a11, false, false);
    buf ^= 1;
  }

  const int mlo = (lane < 16) ? 0 : 8;
  auto storeTile = [&](const v8f& acc, int tmi, int tni) {
    const int   nn  = bn * 64 + wn * 32 + tni * 16 + l15;
    const float bvv = bias ? bias[nn] : 0.f;
    float sc = 1.f, sh = 0.f;
    if constexpr (EPI == 1) {
      sc = bng[nn] * rsqrtf(bnv[nn] + 1e-5f);
      sh = bnb[nn] - bnm[nn] * sc;
    }
#pragma unroll
    for (int r = 0; r < 8; ++r) {
      const int mm = bm * 64 + wm * 32 + tmi * 16 + mlo + r;
      float v = acc[r] + bvv;
      if constexpr (EPI == 1) {
        v = v * sc + sh;
        v = (v < 0.f) ? 0.2f * v : v;
      }
      if constexpr (SMODE == 0) {
        C[(size_t)mm * N + nn] = v;
      } else {
        const int ob = mm / Tout, ot = mm - ob * Tout;
        C[((size_t)ob * N + nn) * Tout + ot] = v;
      }
    }
  };
  storeTile(a00, 0, 0);
  storeTile(a01, 0, 1);
  storeTile(a10, 1, 0);
  storeTile(a11, 1, 1);
}

// ---------------------------------------------------------------------------
// GRU gate fusion: h' = (1-z)*n + z*h  (biases already folded into gi/gh GEMMs)
// ---------------------------------------------------------------------------
__global__ void gru_gate_kernel(const float* __restrict__ gi, long gi_bs,
                                const float* __restrict__ gh,
                                const float* __restrict__ h,
                                float* __restrict__ hout,
                                float* __restrict__ seqout, long seq_bs,
                                int B_, int Hd)
{
  int i = blockIdx.x * blockDim.x + threadIdx.x;
  if (i >= B_ * Hd) return;
  int b = i / Hd, j = i - b * Hd;
  const float* gib = gi + (size_t)b * gi_bs;
  const float* ghb = gh + (size_t)b * 3 * Hd;
  float r = 1.f / (1.f + expf(-(gib[j] + ghb[j])));
  float z = 1.f / (1.f + expf(-(gib[Hd + j] + ghb[Hd + j])));
  float n = tanhf(gib[2 * Hd + j] + r * ghb[2 * Hd + j]);
  float o = (1.f - z) * n + z * h[(size_t)b * Hd + j];
  hout[(size_t)b * Hd + j] = o;
  if (seqout) seqout[(size_t)b * seq_bs + j] = o;
}

// concat(fwd,bwd) -> (B, 2H, T) with batchnorm
__global__ void bn0_kernel(const float* __restrict__ fwd, const float* __restrict__ bwd,
                           const float* __restrict__ g, const float* __restrict__ bb,
                           const float* __restrict__ m, const float* __restrict__ v,
                           float* __restrict__ out, int B_, int T_, int Hd)
{
  long total = (long)B_ * 2 * Hd * T_;
  for (long i = blockIdx.x * (long)blockDim.x + threadIdx.x; i < total;
       i += (long)gridDim.x * blockDim.x) {
    int  t = (int)(i % T_);
    long r = i / T_;
    int  c = (int)(r % (2 * Hd));
    int  b = (int)(r / (2 * Hd));
    float x = (c < Hd) ? fwd[((size_t)b * T_ + t) * Hd + c]
                       : bwd[((size_t)b * T_ + t) * Hd + (c - Hd)];
    out[i] = (x - m[c]) * (g[c] * rsqrtf(v[c] + 1e-5f)) + bb[c];
  }
}

// convT weight prep: w2[co][ci*KW+kk] = w[(ci*Co+co)*KW + (KW-1-kk)]
__global__ void flipw_kernel(const float* __restrict__ w, float* __restrict__ w2,
                             int Ci, int Co, int KW)
{
  long total = (long)Ci * Co * KW;
  for (long i = blockIdx.x * (long)blockDim.x + threadIdx.x; i < total;
       i += (long)gridDim.x * blockDim.x) {
    int co  = (int)(i / ((long)Ci * KW));
    int rem = (int)(i - (long)co * Ci * KW);
    int ci  = rem / KW, kk = rem - ci * KW;
    w2[i] = w[((size_t)ci * Co + co) * KW + (KW - 1 - kk)];
  }
}

__global__ void rownorm_kernel(const float* __restrict__ E, float* __restrict__ out,
                               int n, int d)
{
  int i = blockIdx.x * blockDim.x + threadIdx.x;
  if (i >= n) return;
  const float* r = E + (size_t)i * d;
  float s = 0.f;
  for (int c = 0; c < d; ++c) s += r[c] * r[c];
  out[i] = s;
}

__global__ void zero_kernel(float* __restrict__ p, int n)
{
  int i = blockIdx.x * blockDim.x + threadIdx.x;
  if (i < n) p[i] = 0.f;
}

// decoder input concat: [fb(B,D) | d2[:, :, t] (B,C2) | skel[:, t] (B,D)]
__global__ void pack_cat_kernel(const float* __restrict__ fb, const float* __restrict__ d2,
                                const float* __restrict__ skel, float* __restrict__ cat,
                                int t, int B_, int T_, int D_, int C2, int L2)
{
  const int Wd = 2 * D_ + C2;
  int i = blockIdx.x * blockDim.x + threadIdx.x;
  if (i >= B_ * Wd) return;
  int b = i / Wd, j = i - b * Wd;
  float v;
  if (j < D_)            v = fb ? fb[b * D_ + j] : 0.f;
  else if (j < D_ + C2)  v = d2[((size_t)b * C2 + (j - D_)) * L2 + t];
  else                   v = skel[((size_t)b * T_ + t) * D_ + (j - D_ - C2)];
  cat[i] = v;
}

// logits = h2 @ nfw^T + nfb ; recon write ; fb = (sigmoid - rand > 0)
__global__ void logits_fb_kernel(const float* __restrict__ h2, const float* __restrict__ nfw,
                                 const float* __restrict__ nfb, const float* __restrict__ rnd,
                                 float* __restrict__ recon, float* __restrict__ fb,
                                 int t, int B_, int T_, int D_, int Hd)
{
  int i = blockIdx.x * blockDim.x + threadIdx.x;
  if (i >= B_ * D_) return;
  int b = i / D_, j = i - b * D_;
  const float* hr = h2 + (size_t)b * Hd;
  const float* wr = nfw + (size_t)j * Hd;
  float s = nfb[j];
  for (int k = 0; k < Hd; ++k) s += hr[k] * wr[k];
  recon[((size_t)b * T_ + t) * D_ + j] = s;
  float p = 1.f / (1.f + expf(-s));
  fb[i] = ((p - rnd[((size_t)t * B_ + b) * D_ + j]) > 0.f) ? 1.f : 0.f;
}

// ---------------------------------------------------------------------------
// VQ nearest-neighbor: per block, 16 zt rows kept in WMMA A registers and
// reused across all codebook tiles; argmin with first-index tie break.
// ---------------------------------------------------------------------------
__global__ void __launch_bounds__(128)
vq_kernel(const float* __restrict__ zt, const float* __restrict__ embed,
          const float* __restrict__ enorm, float* __restrict__ qz,
          float* __restrict__ encidx, float* __restrict__ counts,
          float* __restrict__ mparts, int CD_, int NC_, int Tz)
{
  __shared__ float rnorm[16];
  __shared__ float s_v[4][16];
  __shared__ int   s_i[4][16];
  __shared__ int   s_idx[16];
  __shared__ float s_red[128];
  const int tid = threadIdx.x, lane = tid & 31, wv = tid >> 5;
  const int r0  = blockIdx.x * 16;

  if (tid < 16) {
    const float* zr = zt + (size_t)(r0 + tid) * CD_;
    float s = 0.f;
    for (int c = 0; c < CD_; ++c) s += zr[c] * zr[c];
    rnorm[tid] = s;
  }
  __syncthreads();

  const int arow  = r0 + (lane & 15);
  const int kofsA = (lane < 16) ? 0 : 16;
  const int kofsB = (lane < 16) ? 0 : 16;
  v16h a[4];
#pragma unroll
  for (int ks = 0; ks < 4; ++ks)
#pragma unroll
    for (int e = 0; e < 16; ++e) {
      int kk = ks * 32 + kofsA + e;
      a[ks][e] = (_Float16)zt[(size_t)arow * CD_ + kk];
    }

  float minv[8];
  int   mini[8];
#pragma unroll
  for (int r = 0; r < 8; ++r) { minv[r] = 3.4e38f; mini[r] = 0; }

  const int ncl = lane & 15;
  const int mlo = (lane < 16) ? 0 : 8;
  for (int nt = wv; nt < (NC_ >> 4); nt += 4) {
    const int brow = nt * 16 + ncl;
    v8f acc = {};
#pragma unroll
    for (int ks = 0; ks < 4; ++ks) {
      v16h bv;
#pragma unroll
      for (int e = 0; e < 16; ++e)
        bv[e] = (_Float16)embed[(size_t)brow * CD_ + ks * 32 + kofsB + e];
      acc = __builtin_amdgcn_wmma_f32_16x16x32_f16(false, a[ks], false, bv,
                                                   (short)0, acc, false, false);
    }
#pragma unroll
    for (int r = 0; r < 8; ++r) {
      const float d = rnorm[mlo + r] - 2.f * acc[r] + enorm[brow];
      if (d < minv[r]) { minv[r] = d; mini[r] = brow; }  // strictly increasing n in-lane
    }
  }
#pragma unroll
  for (int off = 1; off < 16; off <<= 1) {
#pragma unroll
    for (int r = 0; r < 8; ++r) {
      float ov = __shfl_xor(minv[r], off, 32);
      int   oi = __shfl_xor(mini[r], off, 32);
      if (ov < minv[r] || (ov == minv[r] && oi < mini[r])) { minv[r] = ov; mini[r] = oi; }
    }
  }
  if ((lane & 15) == 0) {
#pragma unroll
    for (int r = 0; r < 8; ++r) { s_v[wv][mlo + r] = minv[r]; s_i[wv][mlo + r] = mini[r]; }
  }
  __syncthreads();
  if (tid < 16) {
    float bvv = s_v[0][tid];
    int   bi  = s_i[0][tid];
    for (int w = 1; w < 4; ++w)
      if (s_v[w][tid] < bvv || (s_v[w][tid] == bvv && s_i[w][tid] < bi)) {
        bvv = s_v[w][tid]; bi = s_i[w][tid];
      }
    s_idx[tid] = bi;
    encidx[r0 + tid] = (float)bi;
    atomicAdd(&counts[bi], 1.0f);  // integer-valued -> order-independent
  }
  __syncthreads();

  float local = 0.f;
  for (int e = tid; e < 16 * CD_; e += blockDim.x) {
    const int m = e / CD_, c = e - m * CD_;
    const int grow = r0 + m;
    const float q  = embed[(size_t)s_idx[m] * CD_ + c];
    const float dz = q - zt[(size_t)grow * CD_ + c];
    local += dz * dz;
    const int b = grow / Tz, tc = grow - b * Tz;
    qz[((size_t)b * CD_ + c) * Tz + tc] = q;  // quant_st == quant numerically
  }
  s_red[tid] = local;
  __syncthreads();
  for (int s = 64; s > 0; s >>= 1) {
    if (tid < s) s_red[tid] += s_red[tid + s];
    __syncthreads();
  }
  if (tid == 0) mparts[blockIdx.x] = s_red[0];  // deterministic block partial
}

__global__ void finalize_kernel(const float* __restrict__ mparts, int np,
                                const float* __restrict__ counts,
                                float* __restrict__ o_loss, float* __restrict__ o_cmt,
                                float* __restrict__ o_perp, float denom, int NC_, float nr)
{
  __shared__ float sh[256];
  int tid = threadIdx.x;
  float e = 0.f;
  for (int i = tid; i < NC_; i += 256) {
    float p = counts[i] / nr;
    e += p * logf(p + 1e-10f);
  }
  sh[tid] = e;
  __syncthreads();
  for (int s = 128; s > 0; s >>= 1) {
    if (tid < s) sh[tid] += sh[tid + s];
    __syncthreads();
  }
  if (tid == 0) {
    float ssum = 0.f;
    for (int i = 0; i < np; ++i) ssum += mparts[i];  // fixed order -> deterministic
    float mse = ssum / denom;
    *o_cmt  = mse;
    *o_loss = mse + 0.2f * mse;  // codebook + 0.2*cmt (numerically equal terms)
    *o_perp = expf(-sh[0]);
  }
}

// ---------------------------------------------------------------------------
// Host orchestration (graph-capture safe: stream-only launches)
// ---------------------------------------------------------------------------
extern "C" void kernel_launch(void* const* d_in, const int* in_sizes, int n_in,
                              void* d_out, int out_size, void* d_ws, size_t ws_size,
                              hipStream_t stream)
{
  (void)in_sizes; (void)n_in; (void)out_size; (void)ws_size;

  const float* note   = (const float*)d_in[0];
  const float* skel   = (const float*)d_in[1];
  const float* skenc  = (const float*)d_in[2];
  const float* ehw    = (const float*)d_in[3];
  const float* ehb    = (const float*)d_in[4];
  const float* ifw    = (const float*)d_in[5];
  const float* ifb    = (const float*)d_in[6];
  const float* gfwih  = (const float*)d_in[7];
  const float* gfwhh  = (const float*)d_in[8];
  const float* gfbih  = (const float*)d_in[9];
  const float* gfbhh  = (const float*)d_in[10];
  const float* gbwih  = (const float*)d_in[11];
  const float* gbwhh  = (const float*)d_in[12];
  const float* gbbih  = (const float*)d_in[13];
  const float* gbbhh  = (const float*)d_in[14];
  const float* bn0g   = (const float*)d_in[15];
  const float* bn0b   = (const float*)d_in[16];
  const float* bn0m   = (const float*)d_in[17];
  const float* bn0v   = (const float*)d_in[18];
  const float* c1w    = (const float*)d_in[19];
  const float* c1b    = (const float*)d_in[20];
  const float* bn1g   = (const float*)d_in[21];
  const float* bn1b   = (const float*)d_in[22];
  const float* bn1m   = (const float*)d_in[23];
  const float* bn1v   = (const float*)d_in[24];
  const float* c2w    = (const float*)d_in[25];
  const float* c2b    = (const float*)d_in[26];
  const float* bn2g   = (const float*)d_in[27];
  const float* bn2b   = (const float*)d_in[28];
  const float* bn2m   = (const float*)d_in[29];
  const float* bn2v   = (const float*)d_in[30];
  const float* c3w    = (const float*)d_in[31];
  const float* c3b    = (const float*)d_in[32];
  const float* embed  = (const float*)d_in[33];
  const float* ct1w   = (const float*)d_in[34];
  const float* ct1b   = (const float*)d_in[35];
  const float* dbn1g  = (const float*)d_in[36];
  const float* dbn1b  = (const float*)d_in[37];
  const float* dbn1m  = (const float*)d_in[38];
  const float* dbn1v  = (const float*)d_in[39];
  const float* ct2w   = (const float*)d_in[40];
  const float* ct2b   = (const float*)d_in[41];
  const float* dbn2g  = (const float*)d_in[42];
  const float* dbn2b  = (const float*)d_in[43];
  const float* dbn2m  = (const float*)d_in[44];
  const float* dbn2v  = (const float*)d_in[45];
  const float* dhw    = (const float*)d_in[46];
  const float* dhb    = (const float*)d_in[47];
  const float* g1wih  = (const float*)d_in[48];
  const float* g1whh  = (const float*)d_in[49];
  const float* g1bih  = (const float*)d_in[50];
  const float* g1bhh  = (const float*)d_in[51];
  const float* g2wih  = (const float*)d_in[52];
  const float* g2whh  = (const float*)d_in[53];
  const float* g2bih  = (const float*)d_in[54];
  const float* g2bhh  = (const float*)d_in[55];
  const float* nfw    = (const float*)d_in[56];
  const float* nfb    = (const float*)d_in[57];
  const float* srand_ = (const float*)d_in[58];

  const int Bc = 128, Tc = 256, Hc = 512, CDc = 128, NCc = 2048, Sc = 256, Dc = 9;
  const int H3 = 3 * Hc;             // 1536
  const int T1 = 128, T2 = 64;       // conv time lengths
  const int L1 = 131, L2 = 264;      // convT time lengths
  const int CATW = 2 * Dc + Hc / 2;  // 274

  // ---- workspace layout (floats); ~412 MB total with aliasing ----
  float* ws   = (float*)d_ws;
  float* x    = ws;                         // 16,777,216  (B*T,H)
  float* gi   = ws + 16777216;              // 50,331,648  (B*T,3H), reused
  float* fwd  = ws + 67108864;              // 16,777,216
  float* bwd  = ws + 83886080;              // 16,777,216
  float* sm   = ws + 100663296;             // small-buffer region
  float* hid  = sm;
  float* hA   = sm + 65536;
  float* hB   = sm + 131072;
  float* gh   = sm + 196608;                // (B,3H)
  float* h1A  = sm + 393216;
  float* h1B  = sm + 458752;
  float* h2A  = sm + 524288;
  float* h2B  = sm + 589824;
  float* cat  = sm + 655360;                // (B,274)
  float* gi1  = sm + 692224;
  float* gh1  = sm + 888832;
  float* gi2  = sm + 1085440;
  float* gh2  = sm + 1282048;
  float* fb   = sm + 1478656;               // (B,D)
  float* w2a  = sm + 1480704;               // 262,144 (512 x 512)
  float* w2b  = sm + 1742848;               // 524,288 (256 x 2048)
  float* enrm = sm + 2267136;               // 2048
  float* cnts = sm + 2269184;               // 2048
  float* mprt = sm + 2271232;               // 512
  // aliases into dead regions
  float* out0 = gi;                         // (B,2H,T)   33,554,432
  float* c1o  = gi + 33554432;              // (B,512,128) 8,388,608
  float* c2o  = gi + 41943040;              // (B,512,64)  4,194,304
  float* ztb  = gi + 46137344;              // (8192,128)  1,048,576
  float* d1   = x;                          // (B,512,131) 8,585,216  (x is dead)
  float* d2   = fwd;                        // (B,256,264) 8,650,752  (fwd is dead)

  // ---- output layout ----
  float* out     = (float*)d_out;
  float* o_recon = out;                     // 294,912
  float* o_qz    = out + 294912;            // 1,048,576
  float* o_loss  = out + 1343488;
  float* o_cmt   = out + 1343489;
  float* o_idx   = out + 1343490;           // 8,192
  float* o_perp  = out + 1351682;

  auto nblk = [](int M, int N) { return (M / 64) * (N / 64); };

  auto gemm = [&](const float* A, const float* W, const float* bias, float* C,
                  int M, int N, int K, int lda) {
    if ((K & 31) == 0)
      wmma_gemm_kernel<0, 0, 0, 1><<<nblk(M, N), 128, 0, stream>>>(
          A, W, bias, C, nullptr, nullptr, nullptr, nullptr,
          M, N, K, lda, 0, 1, 1, 0, 1, 1);
    else
      wmma_gemm_kernel<0, 0, 0, 0><<<nblk(M, N), 128, 0, stream>>>(
          A, W, bias, C, nullptr, nullptr, nullptr, nullptr,
          M, N, K, lda, 0, 1, 1, 0, 1, 1);
  };
  auto gate = [&](const float* gip, long gibs, const float* ghp, const float* hp,
                  float* hn, float* seq, long seqbs) {
    gru_gate_kernel<<<(Bc * Hc + 255) / 256, 256, 0, stream>>>(gip, gibs, ghp, hp, hn,
                                                               seq, seqbs, Bc, Hc);
  };

  // ===================== Encoder =====================
  gemm(skenc, ehw, ehb, hid, Bc, Hc, Sc, Sc);             // hid = lin(skel_encoded)
  gemm(note, ifw, ifb, x, Bc * Tc, Hc, Dc, Dc);           // x = lin(note)  (K=9 guarded)

  // forward GRU
  gemm(x, gfwih, gfbih, gi, Bc * Tc, H3, Hc, Hc);         // batched input projection
  {
    const float* hp = hid; float* hn = hA;
    for (int t = 0; t < Tc; ++t) {
      gemm(hp, gfwhh, gfbhh, gh, Bc, H3, Hc, Hc);
      gate(gi + (size_t)t * H3, (long)Tc * H3, gh, hp, hn,
           fwd + (size_t)t * Hc, (long)Tc * Hc);
      hp = hn; hn = (hn == hA) ? hB : hA;
    }
  }
  // backward GRU
  gemm(x, gbwih, gbbih, gi, Bc * Tc, H3, Hc, Hc);
  {
    const float* hp = hid; float* hn = hA;
    for (int t = Tc - 1; t >= 0; --t) {
      gemm(hp, gbwhh, gbbhh, gh, Bc, H3, Hc, Hc);
      gate(gi + (size_t)t * H3, (long)Tc * H3, gh, hp, hn,
           bwd + (size_t)t * Hc, (long)Tc * Hc);
      hp = hn; hn = (hn == hA) ? hB : hA;
    }
  }

  // concat + bn0 -> (B, 2H, T)
  {
    long total = (long)Bc * 2 * Hc * Tc;
    bn0_kernel<<<(int)((total + 255) / 256), 256, 0, stream>>>(
        fwd, bwd, bn0g, bn0b, bn0m, bn0v, out0, Bc, Tc, Hc);
  }

  // conv stack (BN + leaky fused into GEMM epilogues, async W staging)
  wmma_gemm_kernel<1, 1, 1, 1><<<nblk(Bc * T1, Hc), 128, 0, stream>>>(
      out0, c1w, c1b, c1o, bn1g, bn1b, bn1m, bn1v,
      Bc * T1, Hc, 2 * Hc * 4, 0, 2 * Hc, 4, 2, 1, Tc, T1);
  wmma_gemm_kernel<1, 1, 1, 1><<<nblk(Bc * T2, Hc), 128, 0, stream>>>(
      c1o, c2w, c2b, c2o, bn2g, bn2b, bn2m, bn2v,
      Bc * T2, Hc, Hc * 4, 0, Hc, 4, 2, 1, T1, T2);
  wmma_gemm_kernel<1, 0, 0, 1><<<nblk(Bc * T2, CDc), 128, 0, stream>>>(
      c2o, c3w, c3b, ztb, nullptr, nullptr, nullptr, nullptr,
      Bc * T2, CDc, Hc, 0, Hc, 1, 1, 0, T2, T2);

  // ===================== VQ =====================
  rownorm_kernel<<<(NCc + 255) / 256, 256, 0, stream>>>(embed, enrm, NCc, CDc);
  zero_kernel<<<(2560 + 255) / 256, 256, 0, stream>>>(cnts, 2048 + 512);
  vq_kernel<<<512, 128, 0, stream>>>(ztb, embed, enrm, o_qz, o_idx, cnts, mprt,
                                     CDc, NCc, T2);
  finalize_kernel<<<1, 256, 0, stream>>>(mprt, 512, cnts, o_loss, o_cmt, o_perp,
                                         (float)(Bc * T2 * CDc), NCc, (float)(Bc * T2));

  // ===================== Decoder convT (BN + leaky fused) =====================
  flipw_kernel<<<(262144 + 255) / 256, 256, 0, stream>>>(ct1w, w2a, CDc, Hc, 4);
  flipw_kernel<<<(524288 + 255) / 256, 256, 0, stream>>>(ct2w, w2b, Hc, Hc / 2, 4);
  wmma_gemm_kernel<2, 1, 1, 1><<<nblk(Bc * L1, Hc), 128, 0, stream>>>(
      o_qz, w2a, ct1b, d1, dbn1g, dbn1b, dbn1m, dbn1v,
      Bc * L1, Hc, CDc * 4, 0, CDc, 4, 2, 3, T2, L1);
  wmma_gemm_kernel<2, 1, 1, 1><<<nblk(Bc * L2, Hc / 2), 128, 0, stream>>>(
      d1, w2b, ct2b, d2, dbn2g, dbn2b, dbn2m, dbn2v,
      Bc * L2, Hc / 2, Hc * 4, 0, Hc, 4, 2, 3, L1, L2);

  // ===================== Autoregressive decoder =====================
  gemm(skenc, dhw, dhb, h1A, Bc, Hc, Sc, Sc);  // h1 init
  {
    const float* h1p = h1A; float* h1n = h1B;
    const float* h2p = nullptr; float* h2n = h2A;
    const float* fbp = nullptr;
    for (int t = 0; t < Tc; ++t) {
      pack_cat_kernel<<<(Bc * CATW + 255) / 256, 256, 0, stream>>>(
          fbp, d2, skel, cat, t, Bc, Tc, Dc, Hc / 2, L2);
      gemm(cat, g1wih, g1bih, gi1, Bc, H3, CATW, CATW);
      gemm(h1p, g1whh, g1bhh, gh1, Bc, H3, Hc, Hc);
      gate(gi1, (long)H3, gh1, h1p, h1n, nullptr, 0);
      const float* h2h = (t == 0) ? (const float*)h1n : h2p;  // step0: h2 seeded by h1
      gemm(h1n, g2wih, g2bih, gi2, Bc, H3, Hc, Hc);
      gemm(h2h, g2whh, g2bhh, gh2, Bc, H3, Hc, Hc);
      gate(gi2, (long)H3, gh2, h2h, h2n, nullptr, 0);
      logits_fb_kernel<<<(Bc * Dc + 127) / 128, 128, 0, stream>>>(
          h2n, nfw, nfb, srand_, o_recon, fb, t, Bc, Tc, Dc, Hc);
      fbp = fb;
      h1p = h1n; h1n = (h1n == h1A) ? h1B : h1A;
      h2p = h2n; h2n = (h2n == h2A) ? h2B : h2A;
    }
  }
}
